// NestedSwiGLUMLP_67937792688175
// MI455X (gfx1250) — matrix operand order repaired and checked by reading, not compiled, not run on hardware
//
#include <hip/hip_runtime.h>

#define DD 1024   // in/out features
#define HD 2048   // proj1 width
#define HH 1024   // act width (max)
#define NT 16384  // tokens
#define NE 4      // experts

typedef __attribute__((ext_vector_type(16))) __bf16 v16bf;
typedef __attribute__((ext_vector_type(8)))  __bf16 v8bf;
typedef __attribute__((ext_vector_type(4)))  __bf16 v4bf;
typedef __attribute__((ext_vector_type(8)))  float  v8f;

// ---------------- K0: zero per-expert counters ----------------
__global__ void zero_counts_kernel(int* counts) {
    if (threadIdx.x < NE) counts[threadIdx.x] = 0;
}

// ---------------- K1: group tokens by expert (scatter) ----------------
__global__ void scatter_tokens_kernel(const int* __restrict__ mask,
                                      int* __restrict__ counts,
                                      int* __restrict__ perm) {
    int t = blockIdx.x * blockDim.x + threadIdx.x;
    if (t < NT) {
        int e = mask[t];
        int pos = atomicAdd(&counts[e], 1);
        perm[e * NT + pos] = t;
    }
}

// ---------------- K2: f32 -> bf16 conversion of w1, w2, x (vectorized x4) ----------------
__global__ void convert_bf16_kernel(const float* __restrict__ w1, const float* __restrict__ w2,
                                    const float* __restrict__ x,
                                    __bf16* __restrict__ w1b, __bf16* __restrict__ w2b,
                                    __bf16* __restrict__ xb) {
    const long T = (long)HD * DD;   // 2M per weight
    const long X = (long)NT * DD;   // 16M for x
    long i = ((long)blockIdx.x * blockDim.x + threadIdx.x) * 4;
    const float* src; __bf16* dst; long off;
    if (i < T)            { src = w1; dst = w1b; off = i; }
    else if (i < 2 * T)   { src = w2; dst = w2b; off = i - T; }
    else                  { src = x;  dst = xb;  off = i - 2 * T; if (off >= X) return; }
    float4 v = *(const float4*)(src + off);
    v4bf o; o[0] = (__bf16)v.x; o[1] = (__bf16)v.y; o[2] = (__bf16)v.z; o[3] = (__bf16)v.w;
    *(v4bf*)(dst + off) = o;
}

// A fragment (16x32 bf16, MxK): lane L holds row (L&15); K range
// [k0+16*half, +16) -> two 16B LDS loads.
__device__ __forceinline__ v16bf load_a_frag(const __bf16* p) {
    v8bf lo = *(const v8bf*)p;
    v8bf hi = *(const v8bf*)(p + 8);
    v16bf a;
#pragma unroll
    for (int i = 0; i < 8; ++i) { a[i] = lo[i]; a[8 + i] = hi[i]; }
    return a;
}

__device__ __forceinline__ v8f wmma_bf16(v16bf a, v16bf b, v8f c) {
    return __builtin_amdgcn_wmma_f32_16x16x32_bf16(false, a, false, b, (short)0, c, false, false);
}

// ---------------- K3: GEMM1 + SwiGLU -> act (bf16 workspace) ----------------
// Tile: M=64 tokens (4 subtiles), each wave owns one gate tile + paired up tile.
__global__ __launch_bounds__(256) void gemm1_swiglu_kernel(
    const __bf16* __restrict__ xb, const float* __restrict__ b1,
    const __bf16* __restrict__ w1b,
    const int* __restrict__ counts, const int* __restrict__ perm,
    __bf16* __restrict__ act)
{
    extern __shared__ __bf16 smem[];            // x tile: 64 x (d+8) bf16
    const int e = blockIdx.z;
    const int d = 128 << e;
    if ((int)blockIdx.y >= (1 << e)) return;    // col chunks used: 1,2,4,8
    const int cnt  = counts[e];
    const int row0 = blockIdx.x * 64;
    if (row0 >= cnt) return;

    const int stride = d + 8;
    const int tid  = threadIdx.x;
    const int wv   = tid >> 5, lane = tid & 31;
    const int ln   = lane & 15, hf = lane >> 4;
    const int* permE = perm + e * NT;

    // stage x tile (64 x d) as 16B chunks
    const int cpr = d >> 3;                     // 16B chunks per row
    for (int i = tid; i < 64 * cpr; i += 256) {
        int r  = i >> (4 + e);                  // i / cpr
        int c8 = i & (cpr - 1);
        int rr = row0 + r;
        int t  = permE[rr < cnt ? rr : row0];
        *(v8bf*)(smem + r * stride + c8 * 8) = *(const v8bf*)(xb + (long)t * DD + c8 * 8);
    }
    __syncthreads();

    const int jt   = blockIdx.y * 8 + wv;       // this wave's gate tile
    const int jcol = jt * 16 + ln;
    const __bf16* browg = w1b + (long)jcol * DD;
    const __bf16* browu = w1b + (long)(jcol + d) * DD;

    v8f accg[4] = {}, accu[4] = {};
    for (int k0 = 0; k0 < d; k0 += 32) {
        v16bf bg = *(const v16bf*)(browg + k0 + 16 * hf);
        v16bf bu = *(const v16bf*)(browu + k0 + 16 * hf);
        const __bf16* ap = smem + ln * stride + k0 + 16 * hf;
#pragma unroll
        for (int m = 0; m < 4; ++m) {
            v16bf a = load_a_frag(ap + m * 16 * stride);
            accg[m] = wmma_bf16(a, bg, accg[m]);
            accu[m] = wmma_bf16(a, bu, accu[m]);
        }
    }

    const float bg1 = b1[jcol], bu1 = b1[jcol + d];
#pragma unroll
    for (int m = 0; m < 4; ++m) {
#pragma unroll
        for (int r = 0; r < 8; ++r) {           // C layout: row = r + 8*hf (+16m), col = ln
            int rr = row0 + m * 16 + r + 8 * hf;
            if (rr < cnt) {
                int t = permE[rr];
                float g = accg[m][r] + bg1;
                float u = accu[m][r] + bu1;
                act[(long)t * HH + jcol] = (__bf16)((g / (1.0f + __expf(-g))) * u);
            }
        }
    }
}

// ---------------- K4: GEMM2 -> out (f32, scattered to token rows) ----------------
__global__ __launch_bounds__(256) void gemm2_kernel(
    const __bf16* __restrict__ act, const float* __restrict__ b2,
    const __bf16* __restrict__ w2b,
    const int* __restrict__ counts, const int* __restrict__ perm,
    float* __restrict__ out)
{
    extern __shared__ __bf16 smem[];            // act tile: 64 x (d+8) bf16
    const int e = blockIdx.z;
    const int d = 128 << e;
    if ((int)blockIdx.y >= (1 << e)) return;
    const int cnt  = counts[e];
    const int row0 = blockIdx.x * 64;
    if (row0 >= cnt) return;

    const int stride = d + 8;
    const int tid  = threadIdx.x;
    const int wv   = tid >> 5, lane = tid & 31;
    const int ln   = lane & 15, hf = lane >> 4;
    const int* permE = perm + e * NT;

    // stage act tile (64 x d)
    const int cpr = d >> 3;
    for (int i = tid; i < 64 * cpr; i += 256) {
        int r  = i >> (4 + e);
        int c8 = i & (cpr - 1);
        int rr = row0 + r;
        int t  = permE[rr < cnt ? rr : row0];
        *(v8bf*)(smem + r * stride + c8 * 8) = *(const v8bf*)(act + (long)t * HH + c8 * 8);
    }
    __syncthreads();

    const int icol = (blockIdx.y * 8 + wv) * 16 + ln;
    const __bf16* brow = w2b + (long)icol * HD;

    v8f acc[4] = {};
    for (int k0 = 0; k0 < d; k0 += 32) {
        v16bf b = *(const v16bf*)(brow + k0 + 16 * hf);
        const __bf16* ap = smem + ln * stride + k0 + 16 * hf;
#pragma unroll
        for (int m = 0; m < 4; ++m) {
            v16bf a = load_a_frag(ap + m * 16 * stride);
            acc[m] = wmma_bf16(a, b, acc[m]);
        }
    }

    const float bb = b2[icol];
#pragma unroll
    for (int m = 0; m < 4; ++m) {
#pragma unroll
        for (int r = 0; r < 8; ++r) {
            int rr = row0 + m * 16 + r + 8 * hf;
            if (rr < cnt) {
                int t = permE[rr];
                out[(long)t * DD + icol] = acc[m][r] + bb;
            }
        }
    }

    // nested tail out[:, d:DD) = 0 — only one column-chunk does it
    if (blockIdx.y == 0 && d < DD) {
        for (int r = 0; r < 64; ++r) {
            int rr = row0 + r;
            if (rr >= cnt) break;
            int t = permE[rr];
            for (int c = d + tid; c < DD; c += 256) out[(long)t * DD + c] = 0.0f;
        }
    }
}

extern "C" void kernel_launch(void* const* d_in, const int* in_sizes, int n_in,
                              void* d_out, int out_size, void* d_ws, size_t ws_size,
                              hipStream_t stream) {
    (void)in_sizes; (void)n_in; (void)out_size; (void)ws_size;
    const float* x    = (const float*)d_in[0];
    const float* w1   = (const float*)d_in[1];
    const float* b1   = (const float*)d_in[2];
    const float* w2   = (const float*)d_in[3];
    const float* b2   = (const float*)d_in[4];
    const int*   mask = (const int*)d_in[5];
    float* out = (float*)d_out;

    // ws: [counts->256B] [perm 256KB] [w1b 4MB] [w2b 4MB] [xb 32MB] [act 32MB] = ~72.3MB
    char* ws = (char*)d_ws;
    int*    counts = (int*)ws;
    int*    perm   = (int*)(ws + 256);
    __bf16* w1b    = (__bf16*)(ws + 256 + sizeof(int) * NE * NT);
    __bf16* w2b    = w1b + (size_t)HD * DD;
    __bf16* xb     = w2b + (size_t)HD * DD;
    __bf16* actws  = xb  + (size_t)NT * DD;

    zero_counts_kernel<<<1, 64, 0, stream>>>(counts);
    scatter_tokens_kernel<<<NT / 256, 256, 0, stream>>>(mask, counts, perm);

    const long conv_elems = 2L * HD * DD + (long)NT * DD;          // 20M
    convert_bf16_kernel<<<(int)(conv_elems / 4 / 256), 256, 0, stream>>>(w1, w2, x, w1b, w2b, xb);

    dim3 grid(NT / 64, 8, NE);                                      // oversized; early-exit
    size_t shmem = 64u * (1024u + 8u) * sizeof(__bf16);             // 132,096 B
    gemm1_swiglu_kernel<<<grid, 256, shmem, stream>>>(xb, b1, w1b, counts, perm, actws);
    gemm2_kernel<<<grid, 256, shmem, stream>>>(actws, b2, w2b, counts, perm, out);
}
